// BindingSiteModel_75316546503183
// MI455X (gfx1250) — matrix operand order repaired and checked by reading, not compiled, not run on hardware
//
#include <hip/hip_runtime.h>
#include <hip/hip_bf16.h>
#include <stddef.h>
#include <stdint.h>

// ---------------------------------------------------------------------------
// EGNN binding-site model for MI455X (gfx1250, wave32, WMMA).
//
// - All dense GEMMs -> v_wmma_f32_16x16x32_bf16 (bf16 A/B, f32 accumulate).
// - Weights are pre-packed into the per-lane B-fragment layout so each
//   32x16 B tile is two global_load_b128 per lane (L2-resident, ~240KB/layer).
// - Edge-MLP (273->128->128->128->1) fully fused per 16-edge tile; the
//   h[row]/h[col] gather is staged into LDS with CDNA5 async copies
//   (global_load_async_to_lds_b128, ASYNCcnt) from a bf16 shadow of h.
// - segment_sum scatter via global_atomic_add_f32: agg is 51MB -> lives in
//   the 192MB L2, so the 128M atomics/layer stay on-chip.
// ---------------------------------------------------------------------------

#define NN     100000
#define NE     1000000
#define HID    128
#define EDGENF 16
#define KIN    288   // 273 (2*HID + 1 + EDGENF) padded to multiple of 32
#define KNODE  256   // HID + HID

typedef __attribute__((ext_vector_type(16))) __bf16 v16bf;
typedef __attribute__((ext_vector_type(8)))  __bf16 v8bf;
typedef __attribute__((ext_vector_type(4)))  __bf16 v4bf;
typedef __attribute__((ext_vector_type(8)))  float  v8f;

// ---------------- WMMA fragment helpers (wave32) ----------------

__device__ __forceinline__ v8f wmma_bf16(v16bf a, v16bf b, v8f c) {
  // D = A(16x32 bf16) x B(32x16 bf16) + C(16x16 f32)
  return __builtin_amdgcn_wmma_f32_16x16x32_bf16(
      /*neg_a=*/false, a, /*neg_b=*/false, b,
      /*c_mod=*/(short)0, c, /*reuse_a=*/false, /*reuse_b=*/false);
}

// A fragment: 16x32 tile from a row-major bf16 array (row stride `stride`
// elements, 16B-aligned rows), K offset k0 (multiple of 32).
// ISA 16-bit A layout: lane m holds K k0+8h..k0+8h+7 and k0+8h+16..+23
// (h = lane>>4) -> two contiguous 16B vector loads per lane.
__device__ __forceinline__ v16bf load_a(const __bf16* s, int stride, int k0, int lane) {
  const int m = lane & 15, half = lane >> 4;
  const __bf16* p = s + (size_t)m * stride + k0 + 8 * half;
  const v8bf lo = *(const v8bf*)p;         // K offsets +0..7
  const v8bf hi = *(const v8bf*)(p + 16);  // K offsets +16..23
  return __builtin_shufflevector(lo, hi, 0, 1, 2, 3, 4, 5, 6, 7,
                                 8, 9, 10, 11, 12, 13, 14, 15);
}

// B fragment from pre-packed weights: layout [kt][nt][lane][16 elems],
// 32 contiguous bytes per lane -> two global_load_b128.
__device__ __forceinline__ v16bf load_bp(const __bf16* Wp, int kt, int nt, int lane) {
  return *(const v16bf*)(Wp + ((size_t)((kt * 8 + nt) * 32 + lane)) * 16);
}

__device__ __forceinline__ float silu(float x) {
  return x / (1.0f + __expf(-x));
}

// Async 16-byte global -> LDS copy (CDNA5 TDM-adjacent path, ASYNCcnt).
__device__ __forceinline__ void async_copy16(uint32_t lds_off, const __bf16* gaddr) {
  asm volatile("global_load_async_to_lds_b128 %0, %1, off"
               :: "v"(lds_off), "v"(gaddr) : "memory");
}
__device__ __forceinline__ void wait_async0() {
  asm volatile("s_wait_asynccnt 0x0" ::: "memory");
}

// ---------------- small utility kernels ----------------

// Pack a row-major KxHID f32 weight into bf16 B-fragment layout, K padded
// to Kpad (multiple of 32) with zeros.
// dst index: i = (((kt*8 + nt)*32 + lane)*16 + e)
//   n = nt*16 + (lane&15), k = kt*32 + 8*(lane>>4) + (e&7) + 16*(e>>3)
__global__ void packw_kernel(const float* __restrict__ src, __bf16* __restrict__ dst,
                             int K, int Kpad) {
  const int i = blockIdx.x * blockDim.x + threadIdx.x;
  if (i >= Kpad * HID) return;
  const int e    = i & 15;
  const int lane = (i >> 4) & 31;
  const int nt   = (i >> 9) & 7;
  const int kt   = i >> 12;
  const int n = nt * 16 + (lane & 15);
  const int k = kt * 32 + 8 * (lane >> 4) + (e & 7) + 16 * (e >> 3);
  dst[i] = (k < K) ? (__bf16)src[(size_t)k * HID + n] : (__bf16)0.0f;
}

__global__ void copyf_kernel(const float* __restrict__ src, float* __restrict__ dst, int n) {
  const int i = blockIdx.x * blockDim.x + threadIdx.x;
  if (i < n) dst[i] = src[i];
}

__global__ void zero_layer_kernel(float* __restrict__ agg, float* __restrict__ csum,
                                  float* __restrict__ cnt) {
  const int i = blockIdx.x * blockDim.x + threadIdx.x;
  if (i < NN * HID) agg[i] = 0.0f;
  if (i < NN * 3)   csum[i] = 0.0f;
  if (i < NN)       cnt[i]  = 0.0f;
}

// h0 = emb[argmax(x)] @ in_w + in_b   (IN_NF=4, tiny K -> plain VALU)
__global__ __launch_bounds__(128) void embed_kernel(
    const float* __restrict__ x, const float* __restrict__ emb,
    const float* __restrict__ inw, const float* __restrict__ inb,
    float* __restrict__ h, __bf16* __restrict__ hbf) {
  __shared__ int sIdx;
  const int node = blockIdx.x;
  if (threadIdx.x == 0) {
    const float* xr = x + (size_t)node * 24;
    float best = xr[0]; int bi = 0;
#pragma unroll
    for (int i = 1; i < 24; ++i) { if (xr[i] > best) { best = xr[i]; bi = i; } }
    sIdx = bi;
  }
  __syncthreads();
  const int j = threadIdx.x;
  const float* er = emb + sIdx * 4;
  float v = inb[j];
#pragma unroll
  for (int k = 0; k < 4; ++k) v += er[k] * inw[k * HID + j];
  h[(size_t)node * HID + j]   = v;
  hbf[(size_t)node * HID + j] = (__bf16)v;
}

// ---------------- fused edge-MLP kernel ----------------
// One 128-thread block (4 wave32) per 16-edge tile. Each wave owns two
// 16-wide N tiles (4 waves x 32 = 128 hidden channels).
__global__ __launch_bounds__(128) void edge_kernel(
    const float* __restrict__ coordw, const __bf16* __restrict__ hbf,
    const int* __restrict__ rowi, const int* __restrict__ coli,
    const float* __restrict__ eattr,
    const __bf16* __restrict__ ew1, const float* __restrict__ eb1,
    const __bf16* __restrict__ ew2, const float* __restrict__ eb2,
    const __bf16* __restrict__ cw1, const float* __restrict__ cb1,
    const float* __restrict__ cw2,
    float* __restrict__ agg, float* __restrict__ csum, float* __restrict__ cnt) {
  __shared__ __bf16 sIn[16 * KIN];   // concat [h_row | h_col | radial | eattr | pad]
  __shared__ __bf16 sA[16 * HID];    // activation ping
  __shared__ __bf16 sB[16 * HID];    // activation pong
  __shared__ float  sCd[16 * 3];
  __shared__ float  sW[16];          // radial, later edge weight w
  __shared__ int    sRow[16], sCol[16];

  const int tid  = threadIdx.x;
  const int lane = tid & 31;
  const int wave = tid >> 5;
  const int e0   = blockIdx.x * 16;
  const int nt0  = wave * 2;          // first of two 16-wide N tiles
  const int n0   = wave * 32;
  const int mrow = (lane >> 4) * 8;
  const int nc   = lane & 15;

  if (tid < 16) {
    const int e = e0 + tid;
    const int r = rowi[e], c = coli[e];
    sRow[tid] = r; sCol[tid] = c;
    const float dx = coordw[r * 3 + 0] - coordw[c * 3 + 0];
    const float dy = coordw[r * 3 + 1] - coordw[c * 3 + 1];
    const float dz = coordw[r * 3 + 2] - coordw[c * 3 + 2];
    sCd[tid * 3 + 0] = dx; sCd[tid * 3 + 1] = dy; sCd[tid * 3 + 2] = dz;
    sW[tid] = dx * dx + dy * dy + dz * dz;  // radial
  }
  __syncthreads();

  // Stage h[row] / h[col] (16 edges x 256B each) via async global->LDS
  // copies: 512 chunks of 16B, 4 per thread.
  for (int i = tid; i < 512; i += 128) {
    const int part = i >> 8;            // 0: h_row cols, 1: h_col cols
    const int j = i & 255;
    const int e = j >> 4;
    const int c = (j & 15) * 8;         // bf16 element offset within row
    const int node = part ? sCol[e] : sRow[e];
    const uint32_t lds_off =
        (uint32_t)(uintptr_t)&sIn[e * KIN + part * HID + c];
    async_copy16(lds_off, hbf + (size_t)node * HID + c);
  }
  // Tail columns 256..287: radial, eattr, zero pad (scalar DS stores).
  for (int i = tid; i < 16 * 32; i += 128) {
    const int e = i >> 5, k = 256 + (i & 31);
    __bf16 v;
    if (k == 256)      v = (__bf16)sW[e];
    else if (k < 273)  v = (__bf16)eattr[(size_t)(e0 + e) * EDGENF + (k - 257)];
    else               v = (__bf16)0.0f;
    sIn[e * KIN + k] = v;
  }
  wait_async0();
  __syncthreads();

  // GEMM1: m1 = silu(m_in @ ew1 + eb1), K = 288 (9 k-tiles)
  v8f acc0 = {}, acc1 = {};
#pragma unroll
  for (int kt = 0; kt < KIN / 32; ++kt) {
    const v16bf a = load_a(sIn, KIN, kt * 32, lane);
    acc0 = wmma_bf16(a, load_bp(ew1, kt, nt0, lane), acc0);
    acc1 = wmma_bf16(a, load_bp(ew1, kt, nt0 + 1, lane), acc1);
  }
#pragma unroll
  for (int r = 0; r < 8; ++r) {
    const int M = mrow + r;
    sA[M * HID + n0 + nc]      = (__bf16)silu(acc0[r] + eb1[n0 + nc]);
    sA[M * HID + n0 + 16 + nc] = (__bf16)silu(acc1[r] + eb1[n0 + 16 + nc]);
  }
  __syncthreads();

  // GEMM2: m2 = silu(m1 @ ew2 + eb2), K = 128; scatter agg += m2 (L2 atomics)
  acc0 = (v8f){}; acc1 = (v8f){};
#pragma unroll
  for (int kt = 0; kt < HID / 32; ++kt) {
    const v16bf a = load_a(sA, HID, kt * 32, lane);
    acc0 = wmma_bf16(a, load_bp(ew2, kt, nt0, lane), acc0);
    acc1 = wmma_bf16(a, load_bp(ew2, kt, nt0 + 1, lane), acc1);
  }
#pragma unroll
  for (int r = 0; r < 8; ++r) {
    const int M = mrow + r;
    const float x0 = silu(acc0[r] + eb2[n0 + nc]);
    const float x1 = silu(acc1[r] + eb2[n0 + 16 + nc]);
    sB[M * HID + n0 + nc]      = (__bf16)x0;
    sB[M * HID + n0 + 16 + nc] = (__bf16)x1;
    const int node = sRow[M];
    atomicAdd(&agg[(size_t)node * HID + n0 + nc],      x0);
    atomicAdd(&agg[(size_t)node * HID + n0 + 16 + nc], x1);
  }
  __syncthreads();

  // GEMM3: t = silu(m2 @ cw1 + cb1), K = 128
  acc0 = (v8f){}; acc1 = (v8f){};
#pragma unroll
  for (int kt = 0; kt < HID / 32; ++kt) {
    const v16bf a = load_a(sB, HID, kt * 32, lane);
    acc0 = wmma_bf16(a, load_bp(cw1, kt, nt0, lane), acc0);
    acc1 = wmma_bf16(a, load_bp(cw1, kt, nt0 + 1, lane), acc1);
  }
#pragma unroll
  for (int r = 0; r < 8; ++r) {
    const int M = mrow + r;
    sA[M * HID + n0 + nc]      = (__bf16)silu(acc0[r] + cb1[n0 + nc]);
    sA[M * HID + n0 + 16 + nc] = (__bf16)silu(acc1[r] + cb1[n0 + 16 + nc]);
  }
  __syncthreads();

  // w = t @ cw2 (128 -> 1), then coord scatter: csum += cdiff*w, cnt += 1
  if (tid < 16) {
    float a = 0.0f;
#pragma unroll 8
    for (int k = 0; k < HID; ++k) a += (float)sA[tid * HID + k] * cw2[k];
    sW[tid] = a;
  }
  __syncthreads();
  if (tid < 48) {
    const int e = tid / 3;
    atomicAdd(&csum[(size_t)sRow[e] * 3 + (tid - e * 3)], sCd[tid] * sW[e]);
  }
  if (tid < 16) atomicAdd(&cnt[sRow[tid]], 1.0f);
}

// ---------------- node-MLP kernel (+ coord update) ----------------
__global__ __launch_bounds__(128) void node_kernel(
    float* __restrict__ h, __bf16* __restrict__ hbf, const float* __restrict__ agg,
    const __bf16* __restrict__ nw1, const float* __restrict__ nb1,
    const __bf16* __restrict__ nw2, const float* __restrict__ nb2,
    float* __restrict__ coordw, const float* __restrict__ csum,
    const float* __restrict__ cnt) {
  __shared__ __bf16 sIn[16 * KNODE];  // [h | agg]
  __shared__ __bf16 sA[16 * HID];

  const int tid  = threadIdx.x;
  const int lane = tid & 31;
  const int wave = tid >> 5;
  const int v0   = blockIdx.x * 16;
  const int nt0  = wave * 2;
  const int n0   = wave * 32;
  const int mrow = (lane >> 4) * 8;
  const int nc   = lane & 15;

  if (tid < 48) {  // coord += csum / max(cnt, 1)
    const int i = v0 + tid / 3, d = tid % 3;
    float c = cnt[i]; c = c > 1.0f ? c : 1.0f;
    coordw[(size_t)i * 3 + d] += csum[(size_t)i * 3 + d] / c;
  }

  // Stage [h | agg] as bf16: float4 global loads, packed 8B LDS stores.
  for (int i = tid; i < 16 * KNODE / 4; i += 128) {
    const int m  = i >> 6;
    const int c4 = (i & 63) * 4;
    const float* src = (c4 < HID) ? &h[(size_t)(v0 + m) * HID + c4]
                                  : &agg[(size_t)(v0 + m) * HID + (c4 - HID)];
    const float4 v = *(const float4*)src;
    *(v4bf*)&sIn[m * KNODE + c4] =
        (v4bf){(__bf16)v.x, (__bf16)v.y, (__bf16)v.z, (__bf16)v.w};
  }
  __syncthreads();

  // GEMM1: t = silu([h|agg] @ nw1 + nb1), K = 256
  v8f acc0 = {}, acc1 = {};
#pragma unroll
  for (int kt = 0; kt < KNODE / 32; ++kt) {
    const v16bf a = load_a(sIn, KNODE, kt * 32, lane);
    acc0 = wmma_bf16(a, load_bp(nw1, kt, nt0, lane), acc0);
    acc1 = wmma_bf16(a, load_bp(nw1, kt, nt0 + 1, lane), acc1);
  }
#pragma unroll
  for (int r = 0; r < 8; ++r) {
    const int M = mrow + r;
    sA[M * HID + n0 + nc]      = (__bf16)silu(acc0[r] + nb1[n0 + nc]);
    sA[M * HID + n0 + 16 + nc] = (__bf16)silu(acc1[r] + nb1[n0 + 16 + nc]);
  }
  __syncthreads();

  // GEMM2: h_new = h + (t @ nw2 + nb2), K = 128, in-place residual update
  acc0 = (v8f){}; acc1 = (v8f){};
#pragma unroll
  for (int kt = 0; kt < HID / 32; ++kt) {
    const v16bf a = load_a(sA, HID, kt * 32, lane);
    acc0 = wmma_bf16(a, load_bp(nw2, kt, nt0, lane), acc0);
    acc1 = wmma_bf16(a, load_bp(nw2, kt, nt0 + 1, lane), acc1);
  }
#pragma unroll
  for (int r = 0; r < 8; ++r) {
    const size_t base = (size_t)(v0 + mrow + r) * HID;
    const float x0 = acc0[r] + nb2[n0 + nc]      + h[base + n0 + nc];
    const float x1 = acc1[r] + nb2[n0 + 16 + nc] + h[base + n0 + 16 + nc];
    h[base + n0 + nc] = x0;        hbf[base + n0 + nc] = (__bf16)x0;
    h[base + n0 + 16 + nc] = x1;   hbf[base + n0 + 16 + nc] = (__bf16)x1;
  }
}

// ---------------- output head ----------------
__global__ __launch_bounds__(128) void out_kernel(
    const __bf16* __restrict__ hbf, const __bf16* __restrict__ outw,
    const float* __restrict__ outb, const float* __restrict__ predw,
    const float* __restrict__ predb, float* __restrict__ out) {
  __shared__ float sH[16 * HID];

  const int tid  = threadIdx.x;
  const int lane = tid & 31;
  const int wave = tid >> 5;
  const int v0   = blockIdx.x * 16;
  const int nt0  = wave * 2;
  const int n0   = wave * 32;
  const int mrow = (lane >> 4) * 8;
  const int nc   = lane & 15;

  const __bf16* Abase = hbf + (size_t)v0 * HID;  // A straight from global bf16 h
  v8f acc0 = {}, acc1 = {};
#pragma unroll
  for (int kt = 0; kt < HID / 32; ++kt) {
    const v16bf a = load_a(Abase, HID, kt * 32, lane);
    acc0 = wmma_bf16(a, load_bp(outw, kt, nt0, lane), acc0);
    acc1 = wmma_bf16(a, load_bp(outw, kt, nt0 + 1, lane), acc1);
  }
#pragma unroll
  for (int r = 0; r < 8; ++r) {
    const int M = mrow + r;
    sH[M * HID + n0 + nc]      = acc0[r] + outb[n0 + nc];
    sH[M * HID + n0 + 16 + nc] = acc1[r] + outb[n0 + 16 + nc];
  }
  __syncthreads();
  if (tid < 16) {
    float a = predb[0];
#pragma unroll 8
    for (int k = 0; k < HID; ++k) a += sH[tid * HID + k] * predw[k];
    out[v0 + tid] = 1.0f / (1.0f + __expf(-a));
  }
}

// ---------------------------------------------------------------------------
// Host launcher. Input order assumes JAX sorted-key tree flattening of the
// setup_inputs() dict: x, coord, edge_index, edge_attr, then params leaves:
//   emb, in_b, in_w, layers[0..2]{cb1,cw1,cw2,eb1,eb2,ew1,ew2,nb1,nb2,nw1,nw2},
//   out_b, out_w, pred_b, pred_w  -> 44 tensors total.
// ---------------------------------------------------------------------------
extern "C" void kernel_launch(void* const* d_in, const int* in_sizes, int n_in,
                              void* d_out, int out_size, void* d_ws, size_t ws_size,
                              hipStream_t stream) {
  (void)in_sizes; (void)n_in; (void)out_size; (void)ws_size;

  const float* x     = (const float*)d_in[0];
  const float* coord = (const float*)d_in[1];
  const int*   eidx  = (const int*)d_in[2];
  const float* eattr = (const float*)d_in[3];
  const float* emb   = (const float*)d_in[4];
  const float* inb   = (const float*)d_in[5];
  const float* inw   = (const float*)d_in[6];
  const float* outb  = (const float*)d_in[40];
  const float* outw  = (const float*)d_in[41];
  const float* predb = (const float*)d_in[42];
  const float* predw = (const float*)d_in[43];
  const int* rowp = eidx;
  const int* colp = eidx + NE;

  // workspace carve (~132 MB total)
  size_t off = 0;
  auto carve = [&](size_t bytes) -> char* {
    char* p = (char*)d_ws + off;
    off += (bytes + 255) & ~(size_t)255;
    return p;
  };
  float*  h    = (float*)carve((size_t)NN * HID * 4);
  __bf16* hbf  = (__bf16*)carve((size_t)NN * HID * 2);
  float*  agg  = (float*)carve((size_t)NN * HID * 4);
  float*  csum = (float*)carve((size_t)NN * 3 * 4);
  float*  cnt  = (float*)carve((size_t)NN * 4);
  float*  cw   = (float*)carve((size_t)NN * 3 * 4);  // working coord copy
  __bf16* wE1[3]; __bf16* wE2[3]; __bf16* wC1[3]; __bf16* wN1[3]; __bf16* wN2[3];
  for (int l = 0; l < 3; ++l) {
    wE1[l] = (__bf16*)carve((size_t)KIN * HID * 2);
    wE2[l] = (__bf16*)carve((size_t)HID * HID * 2);
    wC1[l] = (__bf16*)carve((size_t)HID * HID * 2);
    wN1[l] = (__bf16*)carve((size_t)KNODE * HID * 2);
    wN2[l] = (__bf16*)carve((size_t)HID * HID * 2);
  }
  __bf16* wOut = (__bf16*)carve((size_t)HID * HID * 2);

  auto pack = [&](const float* s, __bf16* d, int K, int Kpad) {
    const int n = Kpad * HID;
    packw_kernel<<<(n + 255) / 256, 256, 0, stream>>>(s, d, K, Kpad);
  };
  for (int l = 0; l < 3; ++l) {
    const int b = 7 + 11 * l;  // cb1,cw1,cw2,eb1,eb2,ew1,ew2,nb1,nb2,nw1,nw2
    pack((const float*)d_in[b + 5],  wE1[l], 273, KIN);
    pack((const float*)d_in[b + 6],  wE2[l], HID, HID);
    pack((const float*)d_in[b + 1],  wC1[l], HID, HID);
    pack((const float*)d_in[b + 9],  wN1[l], KNODE, KNODE);
    pack((const float*)d_in[b + 10], wN2[l], HID, HID);
  }
  pack(outw, wOut, HID, HID);

  copyf_kernel<<<(NN * 3 + 255) / 256, 256, 0, stream>>>(coord, cw, NN * 3);
  embed_kernel<<<NN, 128, 0, stream>>>(x, emb, inw, inb, h, hbf);

  for (int l = 0; l < 3; ++l) {
    const int b = 7 + 11 * l;
    zero_layer_kernel<<<(NN * HID + 255) / 256, 256, 0, stream>>>(agg, csum, cnt);
    edge_kernel<<<NE / 16, 128, 0, stream>>>(
        cw, hbf, rowp, colp, eattr,
        wE1[l], (const float*)d_in[b + 3],   // eb1
        wE2[l], (const float*)d_in[b + 4],   // eb2
        wC1[l], (const float*)d_in[b + 0],   // cb1
        (const float*)d_in[b + 2],           // cw2
        agg, csum, cnt);
    node_kernel<<<NN / 16, 128, 0, stream>>>(
        h, hbf, agg,
        wN1[l], (const float*)d_in[b + 7],   // nb1
        wN2[l], (const float*)d_in[b + 8],   // nb2
        cw, csum, cnt);
  }
  out_kernel<<<NN / 16, 128, 0, stream>>>(hbf, wOut, outb, predw, predb, (float*)d_out);
}
